// RankingLoss_83545703842456
// MI455X (gfx1250) — matrix operand order repaired and checked by reading, not compiled, not run on hardware
//
#include <hip/hip_runtime.h>
#include <stdint.h>
#include <stddef.h>

// ---------------- problem constants ----------------
constexpr int BB = 8, HH = 384, WW = 1280;
constexpr int HW = HH * WW;               // 491520
constexpr int NTOT = BB * HW;             // 3932160
constexpr int TILES_Y = HH / 16;          // 24
constexpr int XGRP    = WW / 64;          // 20 (4 tiles of 16 per wave)
constexpr int NPAIRS  = 4915;             // int(1*384*1280*0.01)
constexpr float ONE_P_THETA = 1.15f;

// ---------------- workspace layout (bytes) ----------------
constexpr size_t N_LBLK   = (size_t)NTOT / 256;              // 15360 local-loss blocks
constexpr size_t N_GBLK   = 20;                              // global-loss blocks
constexpr size_t OFF_Q     = 0;                              // 8 f32
constexpr size_t OFF_PLAST = 64;                             // 8 f32
constexpr size_t OFF_PLOSS = 256;                            // N_LBLK f32
constexpr size_t OFF_PCNT  = OFF_PLOSS + N_LBLK * 4;         // N_LBLK f32
constexpr size_t OFF_GLOSS = OFF_PCNT  + N_LBLK * 4;         // N_GBLK f32
constexpr size_t OFF_GCNT  = OFF_GLOSS + N_GBLK * 4;         // N_GBLK f32
constexpr size_t OFF_MK    = 123392;                         // NTOT u8 (valid/invalid bits)
constexpr size_t OFF_GT    = OFF_MK + (size_t)NTOT;          // gt_avg  f32
constexpr size_t OFF_PR    = OFF_GT + (size_t)NTOT * 4;      // pred_avg f32
constexpr size_t OFF_HTG   = OFF_PR + (size_t)NTOT * 4;      // htmp gt  f16
constexpr size_t OFF_HTP   = OFF_HTG + (size_t)NTOT * 2;     // htmp pred f16
constexpr size_t OFF_CUM   = OFF_HTG;                        // cumsum f32 (reuses htmp region after box_v)

// ---------------- vector types ----------------
typedef __attribute__((ext_vector_type(16))) _Float16 v16h;
typedef __attribute__((ext_vector_type(8)))  float    v8f;
typedef __attribute__((ext_vector_type(4)))  uint32_t u32x4;
typedef __attribute__((ext_vector_type(8)))  uint32_t u32x8;

// ---------------- threefry2x32 (JAX-compatible) ----------------
struct TF2 { uint32_t a, b; };

__host__ __device__ inline uint32_t rotl32(uint32_t x, int n) { return (x << n) | (x >> (32 - n)); }

__host__ __device__ inline TF2 threefry2x32(uint32_t k0, uint32_t k1, uint32_t x0, uint32_t x1) {
    uint32_t ks2 = k0 ^ k1 ^ 0x1BD11BDAu;
    x0 += k0; x1 += k1;
    const int R0[4] = {13, 15, 26, 6}, R1[4] = {17, 29, 16, 24};
    for (int r = 0; r < 4; ++r) { x0 += x1; x1 = rotl32(x1, R0[r]); x1 ^= x0; }
    x0 += k1; x1 += ks2 + 1u;
    for (int r = 0; r < 4; ++r) { x0 += x1; x1 = rotl32(x1, R1[r]); x1 ^= x0; }
    x0 += ks2; x1 += k0 + 2u;
    for (int r = 0; r < 4; ++r) { x0 += x1; x1 = rotl32(x1, R0[r]); x1 ^= x0; }
    x0 += k0; x1 += k1 + 3u;
    for (int r = 0; r < 4; ++r) { x0 += x1; x1 = rotl32(x1, R1[r]); x1 ^= x0; }
    x0 += k1; x1 += ks2 + 4u;
    for (int r = 0; r < 4; ++r) { x0 += x1; x1 = rotl32(x1, R0[r]); x1 ^= x0; }
    x0 += ks2; x1 += k0 + 5u;
    return {x0, x1};
}

// JAX random_bits for even-sized draws: counts iota(n) split in halves, pair (i, i+n/2)
__device__ inline uint32_t jax_bits_even(uint32_t k0, uint32_t k1, uint32_t i, uint32_t n) {
    uint32_t h = n >> 1;
    if (i < h) return threefry2x32(k0, k1, i, i + h).a;
    return threefry2x32(k0, k1, i - h, i).b;
}
// n = 4915 (odd): counts padded with one zero, halves of 2458
__device__ inline uint32_t jax_bits_4915(uint32_t k0, uint32_t k1, uint32_t i) {
    const uint32_t h = 2458u;
    if (i < h) { uint32_t c1 = (i == h - 1u) ? 0u : (h + i); return threefry2x32(k0, k1, i, c1).a; }
    return threefry2x32(k0, k1, i - h, i).b;
}
__device__ inline float u01(uint32_t bits) {
    return __uint_as_float(0x3F800000u | (bits >> 9)) - 1.0f;
}
__device__ inline float softplusf(float x) {
    return fmaxf(x, 0.0f) + log1pf(expf(-fabsf(x)));
}

// ---------------- TDM 2D tile load (ISA 08, D# groups 0/1) ----------------
__device__ inline void tdm_load_2d(uint32_t lds_addr, const void* gptr,
                                   uint32_t tensor_d0, uint32_t tensor_d1,
                                   uint32_t tile_d0, uint32_t tile_d1,
                                   uint32_t stride0_elems, uint32_t data_size_code) {
    uint64_t ga = (uint64_t)(uintptr_t)gptr;
    u32x4 g0; u32x8 g1;
    g0[0] = 1u;                                                  // count=1, user descriptor
    g0[1] = lds_addr;                                            // LDS byte address
    g0[2] = (uint32_t)ga;                                        // global_addr[31:0]
    g0[3] = ((uint32_t)(ga >> 32) & 0x1FFFFFFu) | (2u << 30);    // global_addr[56:32], type=2
    g1[0] = (data_size_code & 3u) << 16;                         // wg_mask=0, data_size
    g1[1] = (tensor_d0 & 0xFFFFu) << 16;                         // tensor_dim0[15:0]
    g1[2] = ((tensor_d0 >> 16) & 0xFFFFu) | ((tensor_d1 & 0xFFFFu) << 16);
    g1[3] = ((tensor_d1 >> 16) & 0xFFFFu) | ((tile_d0 & 0xFFFFu) << 16);
    g1[4] = (tile_d1 & 0xFFFFu);                                 // tile_dim1, tile_dim2=0
    g1[5] = stride0_elems;                                       // tensor_dim0_stride lo32
    g1[6] = 0u;                                                  // stride0 hi | stride1 lo
    g1[7] = 0u;
    asm volatile("tensor_load_to_lds %0, %1" :: "s"(g0), "s"(g1) : "memory");
}
__device__ inline void tdm_wait() {
    __builtin_amdgcn_s_wait_tensorcnt(0);
    asm volatile("" ::: "memory");
}

// ---------------- deterministic 256-thread block reduction ----------------
__device__ inline void block_reduce2_256(float a, float b, float* oa, float* ob) {
    __shared__ float sa[256], sb[256];
    int t = threadIdx.x;
    sa[t] = a; sb[t] = b; __syncthreads();
    for (int st = 128; st > 0; st >>= 1) {
        if (t < st) { sa[t] += sa[t + st]; sb[t] += sb[t + st]; }
        __syncthreads();
    }
    if (t == 0) { oa[blockIdx.x] = sa[0]; ob[blockIdx.x] = sb[0]; }
}

// ---------------- K1: per-batch exact 0.95 quantile (radix select) ----------------
__global__ void __launch_bounds__(256) qsel_kernel(const float* __restrict__ mask, float* __restrict__ q) {
    __shared__ uint32_t hist[256];
    __shared__ uint32_t sh_prefix, sh_k, sh_min, sh_cnt;
    const int b = blockIdx.x, tid = threadIdx.x;
    const float idxf = 0.95f * (float)(HW - 1);
    const int   lo   = (int)idxf;
    const float frac = idxf - (float)lo;
    if (tid == 0) { sh_prefix = 0u; sh_k = (uint32_t)lo; }
    __syncthreads();
    for (int p = 3; p >= 0; --p) {
        hist[tid] = 0u;
        __syncthreads();
        const uint32_t prefix = sh_prefix;
        const uint32_t hmask = (p == 3) ? 0u : (0xFFFFFFFFu << ((p + 1) * 8));
        for (int i = tid; i < HW; i += 256) {
            uint32_t u = __float_as_uint(mask[(size_t)b * HW + i]);  // uniforms >= 0: uint order == float order
            if (((u ^ prefix) & hmask) == 0u) atomicAdd(&hist[(u >> (p * 8)) & 255u], 1u);
        }
        __syncthreads();
        if (tid == 0) {
            uint32_t k = sh_k, cum = 0; int sel = 255;
            for (int bin = 0; bin < 256; ++bin) {
                uint32_t c = hist[bin];
                if (cum + c > k) { sel = bin; break; }
                cum += c;
            }
            sh_prefix |= ((uint32_t)sel << (p * 8));
            sh_k = k - cum;
        }
        __syncthreads();
    }
    const uint32_t v1 = sh_prefix;
    if (tid == 0) { sh_min = 0xFFFFFFFFu; sh_cnt = 0u; }
    __syncthreads();
    uint32_t cl = 0, mn = 0xFFFFFFFFu;
    for (int i = tid; i < HW; i += 256) {
        uint32_t u = __float_as_uint(mask[(size_t)b * HW + i]);
        if (u <= v1) cl++; else mn = min(mn, u);
    }
    atomicAdd(&sh_cnt, cl);
    atomicMin(&sh_min, mn);
    __syncthreads();
    if (tid == 0) {
        uint32_t v2 = (sh_cnt > (uint32_t)(lo + 1)) ? v1 : ((sh_min == 0xFFFFFFFFu) ? v1 : sh_min);
        float v1f = __uint_as_float(v1), v2f = __uint_as_float(v2);
        q[b] = v1f + frac * (v2f - v1f);
    }
}

// ---------------- K2: horizontal 5-tap box pass ----------------
// One wave handles 4 adjacent 16x16 output tiles: single TDM stages a 16x80 f32
// region (OOB columns hardware-zero-filled = conv zero padding), then 4 WMMAs
// sharing ONE band-B fragment: out[m][n] = sum_k data[m][16t+k] * band[k][n],
// band[k][n] = 1 iff |(s-64jb) + k - n| <= 2  (sub-tile invariant).
__global__ void __launch_bounds__(32) box_h_kernel(const float* __restrict__ depthV2,
                                                   const float* __restrict__ depth_pred,
                                                   _Float16* __restrict__ ht_gt,
                                                   _Float16* __restrict__ ht_pr) {
    __shared__ float tile[16 * 80];                       // 16 rows x 80-col pitch, f32
    const int jb = blockIdx.x, ti = blockIdx.y;
    const int img = blockIdx.z >> 3, b = blockIdx.z & 7;
    const float*  src = img ? depth_pred : depthV2;
    _Float16*     dst = img ? ht_pr : ht_gt;
    int s = 64 * jb - 2; if (s < 0) s = 0;                // first staged input column
    const float* gsrc = src + (size_t)b * HW + (size_t)(16 * ti) * WW + (size_t)s;
    tdm_load_2d((uint32_t)(uintptr_t)&tile[0], gsrc,
                (uint32_t)(WW - s), 16u,                  // tensor dims: OOB -> zero fill
                80u, 16u,                                 // tile 80 cols x 16 rows
                (uint32_t)WW, 2u);                        // row stride, 4-byte elements
    tdm_wait();

    const int lane = threadIdx.x;
    const int m = lane & 15, hi16 = lane >> 4, n = lane & 15;
    const int soff = s - 64 * jb;                         // 0 (left edge) or -2
    // shared band-B fragment (32x16), generated in registers
    const int kb2 = hi16 * 8;
    v16h bb;
#pragma unroll
    for (int j = 0; j < 16; ++j) {
        int d = soff + kb2 + j + ((j >= 8) ? 8 : 0) - n;
        bb[j] = (d >= -2 && d <= 2) ? (_Float16)1.0f : (_Float16)0.0f;
    }
#pragma unroll
    for (int t = 0; t < 4; ++t) {                         // 4 sub-tiles per wave
        v16h a;                                           // data (16x32 f16), wave32 A layout
#pragma unroll
        for (int v = 0; v < 8; ++v) {
            int kb = ((v >= 4) ? 16 : 0) + (hi16 ? 8 : 0) + 2 * (v & 3);
            float x0 = tile[m * 80 + 16 * t + kb], x1 = tile[m * 80 + 16 * t + kb + 1];
            if (img == 0) {  // where(depthV2 == 0, 1e-5) -- only for real (in-range) columns
                if (x0 == 0.0f && (s + 16 * t + kb)     < WW) x0 = 1e-5f;
                if (x1 == 0.0f && (s + 16 * t + kb + 1) < WW) x1 = 1e-5f;
            }
            a[2 * v]     = (_Float16)x0;
            a[2 * v + 1] = (_Float16)x1;
        }
        v8f c = {};
        c = __builtin_amdgcn_wmma_f32_16x16x32_f16(false, a, false, bb, (short)0, c, false, false);
#pragma unroll
        for (int r = 0; r < 8; ++r) {
            int row = 16 * ti + r + (hi16 ? 8 : 0);
            dst[(size_t)b * HW + (size_t)row * WW + (64 * jb + 16 * t + n)] = (_Float16)c[r];
        }
    }
}

// ---------------- K3: vertical 5-tap box pass, x 1/25 ----------------
// One wave handles 4 adjacent tiles: single TDM stages a 32x64 f16 region
// (128B DMA rows), 4 WMMAs sharing ONE band-A fragment.
__global__ void __launch_bounds__(32) box_v_kernel(const _Float16* __restrict__ ht_gt,
                                                   const _Float16* __restrict__ ht_pr,
                                                   float* __restrict__ gt_avg,
                                                   float* __restrict__ pr_avg) {
    __shared__ _Float16 tile[32 * 64];                    // 32 rows x 64 cols, f16
    const int jb = blockIdx.x, ti = blockIdx.y;
    const int img = blockIdx.z >> 3, b = blockIdx.z & 7;
    const _Float16* src = img ? ht_pr : ht_gt;
    float*          out = img ? pr_avg : gt_avg;
    int s = 16 * ti - 2; if (s < 0) s = 0;                // first staged input row
    const _Float16* gsrc = src + (size_t)b * HW + (size_t)s * WW + (size_t)(64 * jb);
    tdm_load_2d((uint32_t)(uintptr_t)&tile[0], gsrc,
                (uint32_t)(WW - 64 * jb), (uint32_t)(HH - s),
                64u, 32u,                                 // tile 64 cols x 32 rows
                (uint32_t)WW, 1u);                        // row stride, 2-byte elements
    tdm_wait();

    const int lane = threadIdx.x;
    const int m = lane & 15, hi16 = lane >> 4, n = lane & 15;
    // shared band-A fragment (16 out-rows x 32 in-rows)
    v16h a;
#pragma unroll
    for (int j = 0; j < 16; ++j) {
        int v = j >> 1, hlf = j & 1;
        int K = ((v >= 4) ? 16 : 0) + (hi16 ? 8 : 0) + 2 * (v & 3) + hlf;
        int d = (s + K) - (16 * ti + m);
        a[j] = (d >= -2 && d <= 2) ? (_Float16)1.0f : (_Float16)0.0f;
    }
    const int kb2 = hi16 * 8;
#pragma unroll
    for (int t = 0; t < 4; ++t) {                         // 4 sub-tiles per wave
        v16h bb;                                          // data (32x16), wave32 B layout
#pragma unroll
        for (int j = 0; j < 16; ++j) bb[j] = tile[(kb2 + j + ((j >= 8) ? 8 : 0)) * 64 + 16 * t + n];
        v8f c = {};
        c = __builtin_amdgcn_wmma_f32_16x16x32_f16(false, a, false, bb, (short)0, c, false, false);
#pragma unroll
        for (int r = 0; r < 8; ++r) {
            int row = 16 * ti + r + (hi16 ? 8 : 0);
            out[(size_t)b * HW + (size_t)row * WW + (64 * jb + 16 * t + n)] = c[r] * (1.0f / 25.0f);
        }
    }
}

// ---------------- K4: sampling masks (threefry r1/r2 + quantile) ----------------
__global__ void __launch_bounds__(256) mask_kernel(const float* __restrict__ dmask,
                                                   const float* __restrict__ q,
                                                   unsigned char* __restrict__ mk,
                                                   uint32_t r1k0, uint32_t r1k1,
                                                   uint32_t r2k0, uint32_t r2k1) {
    int e = blockIdx.x * 256 + threadIdx.x;
    if (e >= NTOT) return;
    int b = e / HW;
    float r1 = u01(jax_bits_even(r1k0, r1k1, (uint32_t)e, (uint32_t)NTOT));
    float r2 = u01(jax_bits_even(r2k0, r2k1, (uint32_t)e, (uint32_t)NTOT));
    bool itemp = dmask[e] < q[b];
    bool inv0 = (r1 < 0.7f) || itemp;                     // 1 - SAMPLE_RATIO
    unsigned char v = 0;
    if (!inv0) v |= 2;                                    // valid
    if (inv0 && (r2 >= 0.3f)) v |= 1;                     // final invalid (partner pool)
    mk[e] = v;
}

// ---------------- K5: per-batch normalized cumsum (p = inv / max(S,1e-9)) ----------------
__global__ void __launch_bounds__(1024) scan_kernel(const unsigned char* __restrict__ mk,
                                                    float* __restrict__ cum,
                                                    float* __restrict__ plast) {
    __shared__ float sd[1024];
    __shared__ float carry_s, pv_s;
    const int b = blockIdx.x, t = threadIdx.x;
    const unsigned char* mb = mk + (size_t)b * HW;
    float acc = 0.0f;
    for (int i = t; i < HW; i += 1024) acc += (mb[i] & 1) ? 1.0f : 0.0f;
    sd[t] = acc; __syncthreads();
    for (int st = 512; st > 0; st >>= 1) { if (t < st) sd[t] += sd[t + st]; __syncthreads(); }
    if (t == 0) { pv_s = 1.0f / fmaxf(sd[0], 1e-9f); carry_s = 0.0f; }
    __syncthreads();
    const float pv = pv_s;
    float* cb = cum + (size_t)b * HW;
    for (int base = 0; base < HW; base += 1024) {         // HW % 1024 == 0
        float v = (mb[base + t] & 1) ? pv : 0.0f;
        sd[t] = v; __syncthreads();
        for (int off = 1; off < 1024; off <<= 1) {        // inclusive Hillis-Steele
            float u = (t >= off) ? sd[t - off] : 0.0f;
            __syncthreads();
            sd[t] += u; __syncthreads();
        }
        float c0 = carry_s;
        cb[base + t] = c0 + sd[t];
        __syncthreads();
        if (t == 0) carry_s = c0 + sd[1023];
        __syncthreads();
    }
    if (t == 0) plast[b] = carry_s;
}

// ---------------- K6: local ranking loss (partner search + softplus) ----------------
struct Keys16 { uint32_t v[16]; };

__global__ void __launch_bounds__(256) local_loss_kernel(const unsigned char* __restrict__ mk,
                                                         const float* __restrict__ cum,
                                                         const float* __restrict__ plast,
                                                         const float* __restrict__ gta,
                                                         const float* __restrict__ pra,
                                                         float* __restrict__ pl, float* __restrict__ pc,
                                                         Keys16 bk) {
    const int e = blockIdx.x * 256 + threadIdx.x;         // exact grid: NTOT threads
    const int b = e / HW, i = e - b * HW;
    float loss = 0.0f, cnt = 0.0f;
    if (mk[e] & 2) {
        float u = u01(jax_bits_even(bk.v[2 * b], bk.v[2 * b + 1], (uint32_t)i, (uint32_t)HW));
        float r = plast[b] * (1.0f - u);
        const float* cb = cum + (size_t)b * HW;
        int lo = 0, hi = HW;
        while (lo < hi) { int mid = (lo + hi) >> 1; if (cb[mid] < r) lo = mid + 1; else hi = mid; }
        int part = (lo < HW) ? lo : HW - 1;               // searchsorted 'left' + clip
        float za = gta[(size_t)b * HW + part], zb = gta[e];
        float A  = pra[(size_t)b * HW + part], Bv = pra[e];
        float tgt = 0.0f;
        if (za / zb >= ONE_P_THETA) tgt = -1.0f;          // ge_for_mask1=True
        else if (zb / za > ONE_P_THETA) tgt = 1.0f;
        if (tgt != 0.0f) { loss = softplusf(-tgt * (A - Bv)); cnt = 1.0f; }
    }
    block_reduce2_256(loss, cnt, pl, pc);
}

// ---------------- K7: global ranking loss ----------------
__global__ void __launch_bounds__(256) global_loss_kernel(const float* __restrict__ gta,
                                                          const float* __restrict__ pra,
                                                          float* __restrict__ gl, float* __restrict__ gc,
                                                          uint32_t ka0, uint32_t ka1,
                                                          uint32_t kb0, uint32_t kb1) {
    const int t = blockIdx.x * 256 + threadIdx.x;
    float loss = 0.0f, cnt = 0.0f;
    if (t < NPAIRS) {
        int ia = (int)(jax_bits_4915(ka0, ka1, (uint32_t)t) % (uint32_t)HW);
        int ib = (int)(jax_bits_4915(kb0, kb1, (uint32_t)t) % (uint32_t)HW);
        for (int b = 0; b < BB; ++b) {
            float za = gta[(size_t)b * HW + ia], zb = gta[(size_t)b * HW + ib];
            float A  = pra[(size_t)b * HW + ia], Bv = pra[(size_t)b * HW + ib];
            bool ig = (za > 1e-8f) || (zb > 1e-8f);
            float tgt = 0.0f;
            if (za / zb > ONE_P_THETA) tgt = -1.0f;       // strict > for global
            else if (zb / za > ONE_P_THETA) tgt = 1.0f;
            if (ig && tgt != 0.0f) { loss += softplusf(-tgt * (A - Bv)); cnt += 1.0f; }
        }
    }
    block_reduce2_256(loss, cnt, gl, gc);
}

// ---------------- K8: final deterministic reduction ----------------
__global__ void __launch_bounds__(256) final_kernel(const float* __restrict__ pl, const float* __restrict__ pc,
                                                    const float* __restrict__ gl, const float* __restrict__ gc,
                                                    float* __restrict__ out) {
    __shared__ float sa[256], sb[256];
    const int t = threadIdx.x;
    float a = 0.0f, b = 0.0f;
    for (int i = t; i < (int)N_LBLK; i += 256) { a += pl[i]; b += pc[i]; }
    for (int i = t; i < (int)N_GBLK; i += 256) { a += gl[i]; b += gc[i]; }
    sa[t] = a; sb[t] = b; __syncthreads();
    for (int st = 128; st > 0; st >>= 1) { if (t < st) { sa[t] += sa[t + st]; sb[t] += sb[t + st]; } __syncthreads(); }
    if (t == 0) out[0] = sa[0] / sb[0];
}

// ---------------- host: JAX key splits ----------------
static void jax_split(uint32_t k0, uint32_t k1, int num, uint32_t* out /*2*num*/) {
    int n2 = 2 * num, h = num;
    for (int j = 0; j < n2; ++j)
        out[j] = (j < h) ? threefry2x32(k0, k1, (uint32_t)j, (uint32_t)(j + h)).a
                         : threefry2x32(k0, k1, (uint32_t)(j - h), (uint32_t)j).b;
}

extern "C" void kernel_launch(void* const* d_in, const int* in_sizes, int n_in,
                              void* d_out, int out_size, void* d_ws, size_t ws_size,
                              hipStream_t stream) {
    const float* depthV2    = (const float*)d_in[0];
    const float* depth_pred = (const float*)d_in[1];
    const float* depth_mask = (const float*)d_in[2];
    char* ws = (char*)d_ws;

    float*         q     = (float*)(ws + OFF_Q);
    float*         plast = (float*)(ws + OFF_PLAST);
    float*         pl    = (float*)(ws + OFF_PLOSS);
    float*         pc    = (float*)(ws + OFF_PCNT);
    float*         gl    = (float*)(ws + OFF_GLOSS);
    float*         gc    = (float*)(ws + OFF_GCNT);
    unsigned char* mk    = (unsigned char*)(ws + OFF_MK);
    float*         gta   = (float*)(ws + OFF_GT);
    float*         pra   = (float*)(ws + OFF_PR);
    _Float16*      htg   = (_Float16*)(ws + OFF_HTG);
    _Float16*      htp   = (_Float16*)(ws + OFF_HTP);
    float*         cum   = (float*)(ws + OFF_CUM);        // overlays htmp after box_v

    // key(42) -> split 5: k_r1, k_r2, k_part, k_ga, k_gb; then split(k_part, 8)
    uint32_t o10[10];
    jax_split(0u, 42u, 5, o10);
    Keys16 bk;
    jax_split(o10[4], o10[5], 8, bk.v);

    qsel_kernel<<<dim3(BB), dim3(256), 0, stream>>>(depth_mask, q);
    box_h_kernel<<<dim3(XGRP, TILES_Y, 2 * BB), dim3(32), 0, stream>>>(depthV2, depth_pred, htg, htp);
    box_v_kernel<<<dim3(XGRP, TILES_Y, 2 * BB), dim3(32), 0, stream>>>(htg, htp, gta, pra);
    mask_kernel<<<dim3((unsigned)N_LBLK), dim3(256), 0, stream>>>(depth_mask, q, mk, o10[0], o10[1], o10[2], o10[3]);
    scan_kernel<<<dim3(BB), dim3(1024), 0, stream>>>(mk, cum, plast);
    local_loss_kernel<<<dim3((unsigned)N_LBLK), dim3(256), 0, stream>>>(mk, cum, plast, gta, pra, pl, pc, bk);
    global_loss_kernel<<<dim3((unsigned)N_GBLK), dim3(256), 0, stream>>>(gta, pra, gl, gc, o10[6], o10[7], o10[8], o10[9]);
    final_kernel<<<dim3(1), dim3(256), 0, stream>>>(pl, pc, gl, gc, (float*)d_out);
}
